// TraverseSubGNN_32822140076793
// MI455X (gfx1250) — matrix operand order repaired and checked by reading, not compiled, not run on hardware
//
#include <hip/hip_runtime.h>
#include <hip/hip_bf16.h>

// ---------------- problem constants (fixed by setup_inputs) ----------------
#define HEADS 3
#define DIN   256
#define DOUT  192
#define KHEAD 64          // DOUT / HEADS
#define MAXLAYER 3
#define SUBG  500
#define NEG_ATT 0.2f
#define NEG_ACT 0.1f
#define BN_EPS 1e-5f

#define NTILES 12         // DOUT / 16
#define NKB    8          // DIN / 32

typedef __attribute__((ext_vector_type(16))) __bf16 v16bf;
typedef __attribute__((ext_vector_type(8)))  float  v8f;
typedef __attribute__((ext_vector_type(4)))  float  v4f;

union V16BF {
    v16bf v;
    unsigned u[8];
};

// ---- bf16 helpers ----
// RNE conversion (used only in the tiny one-shot W pre-pack kernel)
static __device__ __forceinline__ unsigned f2bf_rne_bits(float f) {
    unsigned u = __float_as_uint(f);
    return (u + 0x7FFFu + ((u >> 16) & 1u)) & 0xFFFF0000u;   // bf16 in TOP 16 bits
}

// pack the top 16 bits of two dwords: result = {hiWord(a1), hiWord(a0)}
// -> single v_perm_b32
static __device__ __forceinline__ unsigned pack_hi16(unsigned a1, unsigned a0) {
    return __builtin_amdgcn_perm(a1, a0, 0x07060302u);
}

// int-trick float atomic max (works for mixed signs, init to -inf)
static __device__ __forceinline__ void atomicMaxFloat(float* addr, float v) {
    if (v >= 0.0f) atomicMax((int*)addr, __float_as_int(v));
    else           atomicMin((unsigned int*)addr, __float_as_uint(v));
}

// ---------------- K0: init scratch ----------------
__global__ void init_kernel(float* __restrict__ aggr, size_t aggrN,
                            float* __restrict__ node_max, float* __restrict__ node_sum, size_t nmN,
                            float* __restrict__ chsum, float* __restrict__ chsumsq) {
    size_t idx = (size_t)blockIdx.x * blockDim.x + threadIdx.x;
    if (idx < aggrN) aggr[idx] = 0.0f;
    if (idx < nmN) {
        node_max[idx] = -__builtin_huge_valf();
        node_sum[idx] = 0.0f;
    }
    if (idx < DOUT) { chsum[idx] = 0.0f; chsumsq[idx] = 0.0f; }
}

// ---------------- K0b: pre-pack W into WMMA B-fragment layout ----------------
// Layout: frag index = (tile*NKB + kb)*32 + lane ; 16 bf16 per frag (32 bytes).
// Element e of lane -> K = kb*32 + (lane>>4)*16 + e ; N = tile*16 + (lane&15).
// hi = RNE bf16 of w ; lo = RNE bf16 of (w - hi).
__global__ void prepack_w_kernel(const float* __restrict__ W,
                                 __bf16* __restrict__ Bhi,
                                 __bf16* __restrict__ Blo) {
    int idx = blockIdx.x * blockDim.x + threadIdx.x;  // (tile*NKB+kb)*32 + lane
    if (idx >= NTILES * NKB * 32) return;
    int lane = idx & 31;
    int tkb  = idx >> 5;
    int kb   = (tkb & (NKB - 1)) * 32;
    int t    = tkb >> 3;
    int n       = lane & 15;
    int halfSel = lane >> 4;
    const float* wcol = W + (size_t)(kb + halfSel * 16) * DOUT + t * 16 + n;

    V16BF hi, lo;
    #pragma unroll
    for (int e = 0; e < 8; ++e) {
        float f0 = wcol[(size_t)(2 * e) * DOUT];
        float f1 = wcol[(size_t)(2 * e + 1) * DOUT];
        unsigned h0 = f2bf_rne_bits(f0);
        unsigned h1 = f2bf_rne_bits(f1);
        hi.u[e] = pack_hi16(h1, h0);
        unsigned l0 = f2bf_rne_bits(f0 - __uint_as_float(h0));
        unsigned l1 = f2bf_rne_bits(f1 - __uint_as_float(h1));
        lo.u[e] = pack_hi16(l1, l0);
    }
    *((v16bf*)Bhi + idx) = hi.v;
    *((v16bf*)Blo + idx) = lo.v;
}

// ---------------- K1: x = subx @ W via bf16 split-precision WMMA ----------------
// Grid: N/16 blocks, 128 threads (4 waves). Wave w owns column tiles {w, w+4, w+8}.
// B fragments come pre-packed (zero conversion VALU in the hot loop); A is split
// on the fly: hi = top16(x) (v_and), lo = exact residual (v_sub), packed with
// v_perm_b32 -> ~3 VALU ops per element, no half-register moves.
__global__ void __launch_bounds__(128) gemm_wmma_kernel(
    const float*  __restrict__ A,     // [N, DIN] fp32
    const __bf16* __restrict__ Bhi,   // packed fragments
    const __bf16* __restrict__ Blo,
    float* __restrict__ X)            // [N, DOUT]
{
    const int lane    = threadIdx.x & 31;
    const int wave    = threadIdx.x >> 5;
    const int rowBase = blockIdx.x * 16;
    const int m       = lane & 15;
    const int halfSel = lane >> 4;

    v8f acc[3] = {};
    const float* arow = A + (size_t)(rowBase + m) * DIN;
    const v16bf* bhiF = (const v16bf*)Bhi;
    const v16bf* bloF = (const v16bf*)Blo;

    #pragma unroll 2
    for (int kbi = 0; kbi < NKB; ++kbi) {
        const int kb = kbi * 32;
        // --- A fragment: contiguous k order, same as the pre-packed B.
        // lane group 0-15 holds K=[kb..kb+16);
        // lane group 16-31 holds K=[kb+16..kb+32).
        const int k0 = kb + halfSel * 16;
        v4f f0 = *(const v4f*)(arow + k0);
        v4f f1 = *(const v4f*)(arow + k0 + 4);
        v4f f2 = *(const v4f*)(arow + k0 + 8);
        v4f f3 = *(const v4f*)(arow + k0 + 12);
        float af[16] = { f0.x,f0.y,f0.z,f0.w, f1.x,f1.y,f1.z,f1.w,
                         f2.x,f2.y,f2.z,f2.w, f3.x,f3.y,f3.z,f3.w };
        V16BF a_hi, a_lo;
        #pragma unroll
        for (int e = 0; e < 8; ++e) {
            float a0 = af[2 * e], a1 = af[2 * e + 1];
            unsigned u0 = __float_as_uint(a0), u1 = __float_as_uint(a1);
            unsigned h0 = u0 & 0xFFFF0000u,  h1 = u1 & 0xFFFF0000u;
            a_hi.u[e] = pack_hi16(u1, u0);                    // v_perm: two bf16 his
            float l0 = a0 - __uint_as_float(h0);              // exact residual
            float l1 = a1 - __uint_as_float(h1);
            a_lo.u[e] = pack_hi16(__float_as_uint(l1), __float_as_uint(l0));
        }

        #pragma unroll
        for (int t = 0; t < 3; ++t) {
            const int colTile = wave + 4 * t;
            const int fidx = (colTile * NKB + kbi) * 32 + lane;
            v16bf b_hi = bhiF[fidx];
            v16bf b_lo = bloF[fidx];
            // split-precision accumulate: hi*hi + hi*lo + lo*hi
            acc[t] = __builtin_amdgcn_wmma_f32_16x16x32_bf16(false, a_hi.v, false, b_hi,
                                                             (short)0, acc[t], false, false);
            acc[t] = __builtin_amdgcn_wmma_f32_16x16x32_bf16(false, a_hi.v, false, b_lo,
                                                             (short)0, acc[t], false, false);
            acc[t] = __builtin_amdgcn_wmma_f32_16x16x32_bf16(false, a_lo.v, false, b_hi,
                                                             (short)0, acc[t], false, false);
        }
    }

    // --- C/D 16x16 f32 layout: VGPR r -> M = r + 8*halfSel ; N = lane&15.
    #pragma unroll
    for (int t = 0; t < 3; ++t) {
        const int colBase = (wave + 4 * t) * 16;
        #pragma unroll
        for (int r = 0; r < 8; ++r) {
            const int mm = r + 8 * halfSel;
            X[(size_t)(rowBase + mm) * DOUT + colBase + m] = acc[t][r];
        }
    }
}

// ---------------- K2: attention logits + per-src max ----------------
__global__ void alpha_kernel(const float* __restrict__ X,
                             const long long* __restrict__ ei,
                             const float* __restrict__ att,
                             float* __restrict__ alpha,
                             float* __restrict__ node_max,
                             int E) {
    int idx = blockIdx.x * blockDim.x + threadIdx.x;
    if (idx >= E * HEADS) return;
    int e = idx / HEADS;
    int h = idx - e * HEADS;
    long long src = ei[e];
    long long dst = ei[(size_t)E + e];
    const float* xi = X + (size_t)dst * DOUT + h * KHEAD;
    const float* xj = X + (size_t)src * DOUT + h * KHEAD;
    const float* at = att + h * 2 * KHEAD;
    float s = 0.0f;
    #pragma unroll 4
    for (int k = 0; k < KHEAD; k += 4) {
        v4f a = *(const v4f*)(xi + k);
        v4f b = *(const v4f*)(at + k);
        s += a.x * b.x + a.y * b.y + a.z * b.z + a.w * b.w;
    }
    #pragma unroll 4
    for (int k = 0; k < KHEAD; k += 4) {
        v4f a = *(const v4f*)(xj + k);
        v4f b = *(const v4f*)(at + KHEAD + k);
        s += a.x * b.x + a.y * b.y + a.z * b.z + a.w * b.w;
    }
    s = (s > 0.0f) ? s : NEG_ATT * s;   // LeakyReLU(0.2)
    alpha[idx] = s;
    atomicMaxFloat(node_max + (size_t)src * HEADS + h, s);
}

// ---------------- K3: exp(alpha - max) and per-src denom ----------------
__global__ void expsum_kernel(const long long* __restrict__ ei,
                              float* __restrict__ alpha,
                              const float* __restrict__ node_max,
                              float* __restrict__ node_sum,
                              int E) {
    int idx = blockIdx.x * blockDim.x + threadIdx.x;
    if (idx >= E * HEADS) return;
    int e = idx / HEADS;
    int h = idx - e * HEADS;
    long long src = ei[e];
    float mx = node_max[(size_t)src * HEADS + h];
    float ex = __expf(alpha[idx] - mx);
    alpha[idx] = ex;                    // overwrite with numerator
    atomicAdd(node_sum + (size_t)src * HEADS + h, ex);
}

// ---------------- K4: masked message scatter (one block per edge) ----------------
__global__ void __launch_bounds__(DOUT) message_kernel(
    const float* __restrict__ X,
    const long long* __restrict__ ei,
    const int* __restrict__ mask,
    const float* __restrict__ alpha,
    const float* __restrict__ node_sum,
    float* __restrict__ aggr,
    int E) {
    int e = blockIdx.x;
    int mk = mask[e];
    if (mk == 0) return;                // masked x_j -> zero message, skip atomics
    int c = threadIdx.x;
    long long src = ei[e];
    long long dst = ei[(size_t)E + e];
    int h = c >> 6;                     // c / KHEAD
    float w = alpha[(size_t)e * HEADS + h] /
              (node_sum[(size_t)src * HEADS + h] + 1e-16f);
    float val = X[(size_t)src * DOUT + c] * w;
    atomicAdd(aggr + (size_t)dst * DOUT + c, val);
}

// ---------------- K5: bias + LeakyReLU(0.1) in-place, per-channel sum/sumsq ----------------
__global__ void __launch_bounds__(DOUT) bnstat_kernel(
    float* __restrict__ aggr,
    const float* __restrict__ bias,
    float* __restrict__ chsum, float* __restrict__ chsumsq,
    int N, int rowsPerBlock) {
    int c = threadIdx.x;
    int r0 = blockIdx.x * rowsPerBlock;
    int r1 = r0 + rowsPerBlock; if (r1 > N) r1 = N;
    float b = bias[c];
    float s = 0.0f, s2 = 0.0f;
    for (int r = r0; r < r1; ++r) {
        float v = aggr[(size_t)r * DOUT + c] + b;
        v = (v > 0.0f) ? v : NEG_ACT * v;
        aggr[(size_t)r * DOUT + c] = v;
        s += v; s2 += v * v;
    }
    atomicAdd(chsum + c, s);
    atomicAdd(chsumsq + c, s2);
}

// ---------------- K5b: finalize BN statistics ----------------
__global__ void bnfinal_kernel(const float* __restrict__ chsum,
                               const float* __restrict__ chsumsq,
                               float* __restrict__ mu, float* __restrict__ rsig,
                               int N) {
    int c = threadIdx.x;
    if (c >= DOUT) return;
    float invN = 1.0f / (float)N;
    float m = chsum[c] * invN;
    float var = chsumsq[c] * invN - m * m;   // biased variance (jnp.var)
    mu[c] = m;
    rsig[c] = rsqrtf(var + BN_EPS);
}

// ---------------- K6: normalize + mean over MAXLAYER axis ----------------
__global__ void out_kernel(const float* __restrict__ aggr,
                           const float* __restrict__ gamma,
                           const float* __restrict__ beta,
                           const float* __restrict__ mu,
                           const float* __restrict__ rsig,
                           float* __restrict__ out,
                           int total) {
    int idx = blockIdx.x * blockDim.x + threadIdx.x;
    if (idx >= total) return;
    int c  = idx % DOUT;
    int bs = idx / DOUT;           // b*SUBG + s
    int s  = bs % SUBG;
    int b  = bs / SUBG;
    float g  = gamma[c] * rsig[c];
    float mm = mu[c];
    float bt = beta[c];
    float acc = 0.0f;
    #pragma unroll
    for (int l = 0; l < MAXLAYER; ++l) {
        int n = (b * MAXLAYER + l) * SUBG + s;
        float v = aggr[(size_t)n * DOUT + c];
        acc += g * (v - mm) + bt;
    }
    out[idx] = acc * (1.0f / MAXLAYER);
}

// ---------------- host-side launcher ----------------
extern "C" void kernel_launch(void* const* d_in, const int* in_sizes, int n_in,
                              void* d_out, int out_size, void* d_ws, size_t ws_size,
                              hipStream_t stream) {
    const float*     subx  = (const float*)d_in[0];
    const long long* ei    = (const long long*)d_in[1];
    const int*       mask  = (const int*)d_in[2];
    const float*     W     = (const float*)d_in[3];
    const float*     att   = (const float*)d_in[4];
    const float*     bias  = (const float*)d_in[5];
    const float*     gamma = (const float*)d_in[6];
    const float*     beta  = (const float*)d_in[7];

    const int N = in_sizes[0] / DIN;   // 96000
    const int E = in_sizes[2];         // 576000 (mask is [E,1])
    float* out = (float*)d_out;

    // scratch carve-out
    char* ws = (char*)d_ws;
    size_t off = 0;
    auto take = [&](size_t nbytes) -> void* {
        void* p = (void*)(ws + off);
        off += (nbytes + 255) & ~(size_t)255;
        return p;
    };
    float*  X       = (float*)take((size_t)N * DOUT * sizeof(float));
    float*  aggr    = (float*)take((size_t)N * DOUT * sizeof(float));
    float*  alphaB  = (float*)take((size_t)E * HEADS * sizeof(float));
    float*  nodeMax = (float*)take((size_t)N * HEADS * sizeof(float));
    float*  nodeSum = (float*)take((size_t)N * HEADS * sizeof(float));
    float*  chsum   = (float*)take(DOUT * sizeof(float));
    float*  chsumsq = (float*)take(DOUT * sizeof(float));
    float*  muB     = (float*)take(DOUT * sizeof(float));
    float*  rsigB   = (float*)take(DOUT * sizeof(float));
    __bf16* Bhi     = (__bf16*)take((size_t)NTILES * NKB * 32 * 16 * sizeof(__bf16));
    __bf16* Blo     = (__bf16*)take((size_t)NTILES * NKB * 32 * 16 * sizeof(__bf16));
    (void)ws_size; (void)n_in;

    // K0: init accumulators (re-run every call for replay determinism)
    {
        size_t aggrN = (size_t)N * DOUT;
        size_t nmN   = (size_t)N * HEADS;
        int blocks = (int)((aggrN + 255) / 256);
        init_kernel<<<blocks, 256, 0, stream>>>(aggr, aggrN, nodeMax, nodeSum, nmN,
                                                chsum, chsumsq);
    }
    // K0b: pre-pack W fragments (tiny; re-run every call, deterministic)
    {
        int total = NTILES * NKB * 32;   // 3072 fragment-threads
        prepack_w_kernel<<<(total + 255) / 256, 256, 0, stream>>>(W, Bhi, Blo);
    }
    // K1: WMMA GEMM
    gemm_wmma_kernel<<<N / 16, 128, 0, stream>>>(subx, Bhi, Blo, X);

    // K2: attention logits + segment max
    {
        int total = E * HEADS;
        alpha_kernel<<<(total + 255) / 256, 256, 0, stream>>>(X, ei, att, alphaB, nodeMax, E);
    }
    // K3: exp + segment sum
    {
        int total = E * HEADS;
        expsum_kernel<<<(total + 255) / 256, 256, 0, stream>>>(ei, alphaB, nodeMax, nodeSum, E);
    }
    // K4: masked message scatter-add
    message_kernel<<<E, DOUT, 0, stream>>>(X, ei, mask, alphaB, nodeSum, aggr, E);

    // K5: bias + LeakyReLU + BN stats
    {
        const int rowsPerBlock = 256;
        int blocks = (N + rowsPerBlock - 1) / rowsPerBlock;
        bnstat_kernel<<<blocks, DOUT, 0, stream>>>(aggr, bias, chsum, chsumsq, N, rowsPerBlock);
    }
    bnfinal_kernel<<<1, DOUT, 0, stream>>>(chsum, chsumsq, muB, rsigB, N);

    // K6: normalize + layer mean
    out_kernel<<<(out_size + 255) / 256, 256, 0, stream>>>(aggr, gamma, beta, muB, rsigB,
                                                           out, out_size);
}
